// SpatialSelfAttention_67053029425436
// MI455X (gfx1250) — compile-verified, matrix operand order checked by reading, not run on hardware
//
#include <hip/hip_runtime.h>
#include <hip/hip_bf16.h>

typedef __attribute__((ext_vector_type(16))) __bf16 v16bf;
typedef __attribute__((ext_vector_type(8)))  __bf16 v8bf;
typedef __attribute__((ext_vector_type(8)))  float  v8f;
typedef unsigned int u32x4 __attribute__((ext_vector_type(4)));
typedef int          i32x8 __attribute__((ext_vector_type(8)));
typedef int          i32x4 __attribute__((ext_vector_type(4)));

#define C_DIM 256
#define S_DIM 4096
#define BATCH 2
#define HEADS 4
#define HDIM  64
#define ROWS  (BATCH * S_DIM)      // 8192
#define N_QKV (3 * C_DIM)          // 768

// ---------------------------------------------------------------------------
// CDNA5 async / tensor data movers. Async loads complete IN ORDER per wave,
// so counted waits implement a 2-deep software pipeline.
// ---------------------------------------------------------------------------
#define WAIT_ASYNC(n)  asm volatile("s_wait_asynccnt " #n ::: "memory")
#define WAIT_TENSOR(n) asm volatile("s_wait_tensorcnt " #n ::: "memory")

__device__ __forceinline__ void async_load_b128(void* lds_ptr, const void* gptr) {
    // LDS aperture lives entirely in addr[63:32]; low 32 bits are the LDS offset.
    unsigned lo = (unsigned)(uintptr_t)lds_ptr;
    asm volatile("global_load_async_to_lds_b128 %0, %1, off"
                 :: "v"(lo), "v"(gptr) : "memory");
}

// TDM 2D load: tile (tile1 rows x tile0 contiguous elems) of a bf16 tensor
// (row stride stride0 elements) into LDS at lds_addr (packed row-major, ld=tile0).
// D# packing per CDNA5 ISA ch.8 (group0: count/lds/global/type, group1: dims).
__device__ __forceinline__ void tdm_load_2d(void* lds_ptr, const void* gptr,
                                            unsigned dim0, unsigned dim1,
                                            unsigned stride0,
                                            unsigned tile0, unsigned tile1) {
    unsigned long long ga = (unsigned long long)(uintptr_t)gptr;
    u32x4 g0;
    g0[0] = 1u;                                        // count=1, user D#
    g0[1] = (unsigned)(uintptr_t)lds_ptr;              // lds_addr (bytes)
    g0[2] = (unsigned)(ga & 0xffffffffu);              // global_addr[31:0]
    g0[3] = (unsigned)((ga >> 32) & 0x01ffffffu)       // global_addr[56:32]
            | 0x80000000u;                             // type=2 ("image")
    i32x8 g1;
    g1[0] = 0x00010000;                                // data_size=1 (2 bytes)
    g1[1] = (int)(dim0 << 16);                         // tensor_dim0[15:0]
    g1[2] = (int)((dim0 >> 16) | (dim1 << 16));        // dim0 hi / dim1 lo
    g1[3] = (int)((dim1 >> 16) | (tile0 << 16));       // dim1 hi / tile_dim0
    g1[4] = (int)(tile1 & 0xffffu);                    // tile_dim1 (tile_dim2=0)
    g1[5] = (int)stride0;                              // tensor_dim0_stride lo32
    g1[6] = 0;
    g1[7] = 0;
    i32x4 z4 = {0, 0, 0, 0};
    i32x8 z8 = {0, 0, 0, 0, 0, 0, 0, 0};
    __builtin_amdgcn_tensor_load_to_lds(g0, g1, z4, z4, z8, 0);
}

// ---------------------------------------------------------------------------
// 16x32 bf16 operand fragment (A-style layout, CDNA5 ISA 7.12.2)
// ---------------------------------------------------------------------------
__device__ __forceinline__ v16bf load_frag(const __bf16* base, int ld, int lane) {
    const int row = lane & 15;
    const int kb  = (lane >> 4) << 3;
    const __bf16* p = base + row * ld + kb;
    v8bf lo = *(const v8bf*)(p);
    v8bf hi = *(const v8bf*)(p + 16);
    v16bf f;
#pragma unroll
    for (int i = 0; i < 8; ++i) { f[i] = lo[i]; f[i + 8] = hi[i]; }
    return f;
}

__device__ __forceinline__ v8f wmma_bf16(v16bf a, v16bf b, v8f c) {
    return __builtin_amdgcn_wmma_f32_16x16x32_bf16(false, a, false, b,
                                                   (short)0, c, false, false);
}

// ---------------------------------------------------------------------------
// Kernel 1a: x (b,C,S) f32  ->  xt (b,S,C) bf16
// ---------------------------------------------------------------------------
__global__ void pack_x_kernel(const float* __restrict__ x, __bf16* __restrict__ xt) {
    int idx = blockIdx.x * blockDim.x + threadIdx.x;   // (b, s, c) flat
    if (idx >= BATCH * S_DIM * C_DIM) return;
    int c = idx & (C_DIM - 1);
    int s = (idx >> 8) & (S_DIM - 1);
    int b = idx >> 20;
    xt[idx] = (__bf16)x[((size_t)(b * C_DIM + c)) * S_DIM + s];
}

// Kernel 1b: weights f32 -> bf16
__global__ void pack_w_kernel(const float* __restrict__ wqkv, const float* __restrict__ wout,
                              __bf16* __restrict__ wq_b, __bf16* __restrict__ wo_b) {
    int idx = blockIdx.x * blockDim.x + threadIdx.x;
    if (idx < N_QKV * C_DIM) {
        wq_b[idx] = (__bf16)wqkv[idx];
    } else {
        int j = idx - N_QKV * C_DIM;
        if (j < C_DIM * C_DIM) wo_b[j] = (__bf16)wout[j];
    }
}

// ---------------------------------------------------------------------------
// Kernel 2: QKV GEMM (8192 x 768 x 256). 64x64 tile / 8 waves.
// Weight slice (64x256, 32 KB) pulled once per block by TDM; A tiles streamed
// with a 2-deep async pipeline (counted s_wait_asynccnt).
// ---------------------------------------------------------------------------
__global__ __launch_bounds__(256)
void qkv_gemm_kernel(const __bf16* __restrict__ xt, const __bf16* __restrict__ w,
                     const float* __restrict__ bias,
                     __bf16* __restrict__ qb, __bf16* __restrict__ kb,
                     __bf16* __restrict__ vtb) {
    __shared__ __align__(16) __bf16 lds_w[64 * 256];      // whole B^T slice, 32 KB
    __shared__ __align__(16) __bf16 lds_a[2][64 * 32];    // double-buffered A
    const int tid  = threadIdx.x;
    const int lane = tid & 31;
    const int wave = tid >> 5;
    const int wr = wave >> 2, wc = wave & 3;
    const int mb = blockIdx.x * 64;
    const int nb = blockIdx.y * 64;

    if (wave == 0)   // one TDM per workgroup: rows nb..nb+63 of w (768x256)
        tdm_load_2d(lds_w, w + (size_t)nb * C_DIM, C_DIM, 64, C_DIM, C_DIM, 64);
    WAIT_TENSOR(0);
    __syncthreads();

    auto issueA = [&](int kk, int buf) {   // 64x32 tile: 1 chunk / thread
        int row = tid >> 2, off = (tid & 3) << 3;
        async_load_b128(&lds_a[buf][row * 32 + off],
                        &xt[(size_t)(mb + row) * C_DIM + kk + off]);
    };

    v8f accs[2] = {};
    issueA(0, 0);
    const int NIT = C_DIM / 32;                      // 8
    for (int i = 0; i < NIT; ++i) {
        const int cur = i & 1;
        if (i + 1 < NIT) { issueA((i + 1) * 32, cur ^ 1); WAIT_ASYNC(1); }
        else             { WAIT_ASYNC(0); }
        __syncthreads();
        v16bf bfrag = load_frag(&lds_w[(wc * 16) * C_DIM + i * 32], C_DIM, lane);
        v16bf a0 = load_frag(&lds_a[cur][(wr * 2 + 0) * 16 * 32], 32, lane);
        v16bf a1 = load_frag(&lds_a[cur][(wr * 2 + 1) * 16 * 32], 32, lane);
        accs[0] = wmma_bf16(a0, bfrag, accs[0]);
        accs[1] = wmma_bf16(a1, bfrag, accs[1]);
        __syncthreads();
    }
#pragma unroll
    for (int t = 0; t < 2; ++t) {
#pragma unroll
        for (int r = 0; r < 8; ++r) {
            int ml = (wr * 2 + t) * 16 + r + ((lane >> 4) << 3);
            int nl = wc * 16 + (lane & 15);
            int m = mb + ml, n = nb + nl;
            float val = accs[t][r] + bias[n];
            int s = m & (S_DIM - 1), bi = m >> 12;
            int blk = n >> 8, c = n & 255, h = c >> 6, dd = c & 63;
            size_t bh = (size_t)(bi * HEADS + h);
            if (blk == 0)
                qb[(bh * S_DIM + s) * HDIM + dd] = (__bf16)(val * 0.125f); // 1/sqrt(64)
            else if (blk == 1)
                kb[(bh * S_DIM + s) * HDIM + dd] = (__bf16)val;
            else
                vtb[(bh * HDIM + dd) * S_DIM + s] = (__bf16)val;          // V^T
        }
    }
}

// ---------------------------------------------------------------------------
// Kernel 3: flash attention. 128 threads = 4 waves; each wave owns 16 queries.
// K / V^T tiles streamed through a 2-deep async pipeline (4 chunks/thread/tile,
// so the counted wait is s_wait_asynccnt 4 while the next tile is in flight).
// ---------------------------------------------------------------------------
__global__ __launch_bounds__(128)
void attn_kernel(const __bf16* __restrict__ q, const __bf16* __restrict__ k,
                 const __bf16* __restrict__ vt, __bf16* __restrict__ attn) {
    __shared__ __align__(16) __bf16 lds_k[2][32 * 64];     // 32 keys x 64 d
    __shared__ __align__(16) __bf16 lds_v[2][64 * 32];     // V^T tile
    __shared__ __align__(16) __bf16 lds_p[4][16 * 32];     // per-wave P tile
    const int tid = threadIdx.x, lane = tid & 31, wave = tid >> 5;
    const int bh = blockIdx.y;
    const int b = bh >> 2, h = bh & 3;
    const __bf16* qb = q  + (size_t)bh * S_DIM * HDIM;
    const __bf16* kp = k  + (size_t)bh * S_DIM * HDIM;
    const __bf16* vp = vt + (size_t)bh * HDIM * S_DIM;
    const int q0 = blockIdx.x * 64 + wave * 16;

    v16bf qa0 = load_frag(qb + (size_t)q0 * HDIM,      HDIM, lane);
    v16bf qa1 = load_frag(qb + (size_t)q0 * HDIM + 32, HDIM, lane);

    auto issue_kv = [&](int kk, int buf) {   // 4 chunks / thread
#pragma unroll
        for (int it = 0; it < 2; ++it) {
            int c = tid + it * 128;
            { int row = c >> 3, off = (c & 7) << 3;   // K tile 32x64
              async_load_b128(&lds_k[buf][row * 64 + off],
                              &kp[(size_t)(kk + row) * HDIM + off]); }
            { int row = c >> 2, off = (c & 3) << 3;   // V^T tile 64x32
              async_load_b128(&lds_v[buf][row * 32 + off],
                              &vp[(size_t)row * S_DIM + kk + off]); }
        }
    };

    float mrow[8], lrow[8];
#pragma unroll
    for (int r = 0; r < 8; ++r) { mrow[r] = -1e30f; lrow[r] = 0.f; }
    v8f oacc[4] = {};

    issue_kv(0, 0);
    const int NIT = S_DIM / 32;                      // 128
    for (int i = 0; i < NIT; ++i) {
        const int cur = i & 1;
        if (i + 1 < NIT) { issue_kv((i + 1) * 32, cur ^ 1); WAIT_ASYNC(4); }
        else             { WAIT_ASYNC(0); }
        __syncthreads();

        // scores: 16 queries x 32 keys, K-loop over d (2 steps of 32)
        v8f s0 = {}, s1 = {};
#pragma unroll
        for (int dk = 0; dk < 2; ++dk) {
            v16bf aa = dk ? qa1 : qa0;
            v16bf b0 = load_frag(&lds_k[cur][0 * 16 * 64 + dk * 32], 64, lane);
            v16bf b1 = load_frag(&lds_k[cur][1 * 16 * 64 + dk * 32], 64, lane);
            s0 = wmma_bf16(aa, b0, s0);
            s1 = wmma_bf16(aa, b1, s1);
        }

        // online softmax (f32); row r lives in 16-lane group g = lane>>4
        float p0[8], p1[8];
#pragma unroll
        for (int r = 0; r < 8; ++r) {
            float mx = fmaxf(s0[r], s1[r]);
#pragma unroll
            for (int m2 = 1; m2 <= 8; m2 <<= 1) mx = fmaxf(mx, __shfl_xor(mx, m2, 32));
            float mn = fmaxf(mrow[r], mx);
            float corr = __expf(mrow[r] - mn);
            float e0 = __expf(s0[r] - mn);
            float e1 = __expf(s1[r] - mn);
            float sum = e0 + e1;
#pragma unroll
            for (int m2 = 1; m2 <= 8; m2 <<= 1) sum += __shfl_xor(sum, m2, 32);
            lrow[r] = lrow[r] * corr + sum;
            mrow[r] = mn;
            p0[r] = e0; p1[r] = e1;
#pragma unroll
            for (int f = 0; f < 4; ++f) oacc[f][r] = oacc[f][r] * corr;
        }

        // route P through per-wave LDS to get the A-fragment layout
#pragma unroll
        for (int r = 0; r < 8; ++r) {
            int row = r + ((lane >> 4) << 3);
            lds_p[wave][row * 32 + (lane & 15)]      = (__bf16)p0[r];
            lds_p[wave][row * 32 + 16 + (lane & 15)] = (__bf16)p1[r];
        }
        v16bf pa = load_frag(&lds_p[wave][0], 32, lane);
#pragma unroll
        for (int dt = 0; dt < 4; ++dt) {
            v16bf bv = load_frag(&lds_v[cur][dt * 16 * 32], 32, lane);
            oacc[dt] = wmma_bf16(pa, bv, oacc[dt]);
        }
        __syncthreads();   // readers done before buffer reuse
    }

#pragma unroll
    for (int dt = 0; dt < 4; ++dt)
#pragma unroll
        for (int r = 0; r < 8; ++r) {
            int row = r + ((lane >> 4) << 3);
            int s = q0 + row;
            int cc = h * HDIM + dt * 16 + (lane & 15);
            float val = oacc[dt][r] / lrow[r];
            attn[((size_t)(b * S_DIM + s)) * C_DIM + cc] = (__bf16)val;
        }
}

// ---------------------------------------------------------------------------
// Kernel 4: out-projection (8192 x 256 x 256) fused with LayerNorm over C.
// B^T tile (256x32, 16 KB) per k-step via pipelined TDM; A tile via pipelined
// async loads. Deterministic 2-stage LDS reduction; writes (b, C, S) f32.
// ---------------------------------------------------------------------------
__global__ __launch_bounds__(256)
void proj_ln_kernel(const __bf16* __restrict__ attn, const __bf16* __restrict__ w,
                    const float* __restrict__ bias, const float* __restrict__ g,
                    const float* __restrict__ bb, float* __restrict__ out) {
    __shared__ __align__(16) __bf16 lds_a[2][32 * 32];
    __shared__ __align__(16) __bf16 lds_b[2][256 * 32];
    __shared__ float red_s[8][32], red_q[8][32];
    __shared__ float s_mu[32], s_inv[32];
    const int tid = threadIdx.x, lane = tid & 31, wave = tid >> 5;
    const int mb = blockIdx.x * 32;

    auto issue_ab = [&](int kk, int buf) {
        if (tid < 128) {   // A tile 32x32: 128 chunks (waves 0-3, uniform)
            int row = tid >> 2, off = (tid & 3) << 3;
            async_load_b128(&lds_a[buf][row * 32 + off],
                            &attn[(size_t)(mb + row) * C_DIM + kk + off]);
        }
        if (wave == 0)     // B^T tile: 256 rows x 32 contiguous, stride 256
            tdm_load_2d(lds_b[buf], w + kk, 32, C_DIM, C_DIM, 32, C_DIM);
    };

    v8f acc[2][2] = {};
    issue_ab(0, 0);
    const int NIT = C_DIM / 32;                      // 8
    for (int i = 0; i < NIT; ++i) {
        const int cur = i & 1;
        if (i + 1 < NIT) { issue_ab((i + 1) * 32, cur ^ 1); WAIT_ASYNC(1); WAIT_TENSOR(1); }
        else             { WAIT_ASYNC(0); WAIT_TENSOR(0); }
        __syncthreads();
        v16bf a0 = load_frag(&lds_a[cur][0],       32, lane);
        v16bf a1 = load_frag(&lds_a[cur][16 * 32], 32, lane);
#pragma unroll
        for (int nt = 0; nt < 2; ++nt) {
            v16bf bf = load_frag(&lds_b[cur][(wave * 32 + nt * 16) * 32], 32, lane);
            acc[0][nt] = wmma_bf16(a0, bf, acc[0][nt]);
            acc[1][nt] = wmma_bf16(a1, bf, acc[1][nt]);
        }
        __syncthreads();
    }

    // bias add + per-row partial sums (each wave covers 32 of the 256 cols)
#pragma unroll
    for (int mt = 0; mt < 2; ++mt) {
#pragma unroll
        for (int r = 0; r < 8; ++r) {
            int row = mt * 16 + r + ((lane >> 4) << 3);
            float part = 0.f, partq = 0.f;
#pragma unroll
            for (int nt = 0; nt < 2; ++nt) {
                int n = wave * 32 + nt * 16 + (lane & 15);
                float v = acc[mt][nt][r] + bias[n];
                acc[mt][nt][r] = v;
                part += v; partq += v * v;
            }
#pragma unroll
            for (int m2 = 1; m2 <= 8; m2 <<= 1) {
                part  += __shfl_xor(part,  m2, 32);
                partq += __shfl_xor(partq, m2, 32);
            }
            if ((lane & 15) == 0) { red_s[wave][row] = part; red_q[wave][row] = partq; }
        }
    }
    __syncthreads();
    if (tid < 32) {          // deterministic cross-wave reduction
        float s = 0.f, q2 = 0.f;
#pragma unroll
        for (int w2 = 0; w2 < 8; ++w2) { s += red_s[w2][tid]; q2 += red_q[w2][tid]; }
        float mu  = s * (1.f / 256.f);
        float var = q2 * (1.f / 256.f) - mu * mu;
        s_mu[tid]  = mu;
        s_inv[tid] = rsqrtf(var + 1e-5f);
    }
    __syncthreads();

#pragma unroll
    for (int mt = 0; mt < 2; ++mt)
#pragma unroll
        for (int r = 0; r < 8; ++r) {
            int row = mt * 16 + r + ((lane >> 4) << 3);
            int m = mb + row;
            int bi = m >> 12, s = m & (S_DIM - 1);
            float mu = s_mu[row], inv = s_inv[row];
#pragma unroll
            for (int nt = 0; nt < 2; ++nt) {
                int n = wave * 32 + nt * 16 + (lane & 15);
                float y = (acc[mt][nt][r] - mu) * inv * g[n] + bb[n];
                out[((size_t)(bi * C_DIM + n)) * S_DIM + s] = y;   // (b, C, S)
            }
        }
}

// ---------------------------------------------------------------------------
extern "C" void kernel_launch(void* const* d_in, const int* in_sizes, int n_in,
                              void* d_out, int out_size, void* d_ws, size_t ws_size,
                              hipStream_t stream) {
    const float* x    = (const float*)d_in[0];
    const float* wqkv = (const float*)d_in[1];
    const float* bqkv = (const float*)d_in[2];
    const float* wout = (const float*)d_in[3];
    const float* bout = (const float*)d_in[4];
    const float* lng  = (const float*)d_in[5];
    const float* lnb  = (const float*)d_in[6];
    float* out = (float*)d_out;

    char* ws = (char*)d_ws;
    size_t off = 0;
    auto alloc = [&](size_t bytes) {
        char* p = ws + off;
        off += (bytes + 255) & ~(size_t)255;
        return p;
    };
    __bf16* xt  = (__bf16*)alloc((size_t)ROWS * C_DIM * 2);
    __bf16* wqb = (__bf16*)alloc((size_t)N_QKV * C_DIM * 2);
    __bf16* wob = (__bf16*)alloc((size_t)C_DIM * C_DIM * 2);
    __bf16* qb  = (__bf16*)alloc((size_t)BATCH * HEADS * S_DIM * HDIM * 2);
    __bf16* kbf = (__bf16*)alloc((size_t)BATCH * HEADS * S_DIM * HDIM * 2);
    __bf16* vtb = (__bf16*)alloc((size_t)BATCH * HEADS * HDIM * S_DIM * 2);
    __bf16* att = (__bf16*)alloc((size_t)ROWS * C_DIM * 2);

    pack_x_kernel<<<(BATCH * S_DIM * C_DIM) / 256, 256, 0, stream>>>(x, xt);
    pack_w_kernel<<<(N_QKV * C_DIM + C_DIM * C_DIM) / 256, 256, 0, stream>>>(
        wqkv, wout, wqb, wob);
    qkv_gemm_kernel<<<dim3(ROWS / 64, N_QKV / 64), 256, 0, stream>>>(
        xt, wqb, bqkv, qb, kbf, vtb);
    attn_kernel<<<dim3(S_DIM / 64, BATCH * HEADS), 128, 0, stream>>>(
        qb, kbf, vtb, att);
    proj_ln_kernel<<<ROWS / 32, 256, 0, stream>>>(att, wob, bout, lng, lnb, out);
}
